// Bert_47090021433692
// MI455X (gfx1250) — compile-verified, matrix operand order checked by reading, not run on hardware
//
#include <hip/hip_runtime.h>
#include <hip/hip_bf16.h>
#include <math.h>

// ---------------------------------------------------------------------------
// DeBERTa-style encoder forward on gfx1250 (MI455X), bf16 WMMA everywhere.
// S=512 B=16 H=768 NH=12 HD=64 I=2048 L=12
// GEMM staging uses gfx1250 async global->LDS copies (ASYNCcnt) with
// double-buffered LDS so v_wmma overlaps the next tile's fill.
// ---------------------------------------------------------------------------

typedef __bf16 bf16;
typedef __attribute__((ext_vector_type(16))) __bf16 v16bf;
typedef __attribute__((ext_vector_type(8)))  __bf16 v8bf;
typedef __attribute__((ext_vector_type(8)))  float  v8f;

#define S_   512
#define B_   16
#define H_   768
#define NH_  12
#define HD_  64
#define L_   12
#define I2_  4096
#define II_  2048
#define SCALE_ 0.07216878364870323f   /* 1/sqrt(3*64) */

__device__ __forceinline__ bf16  f2b(float f) { return (bf16)f; }
__device__ __forceinline__ float b2f(bf16 b)  { return (float)b; }

#define WMMA(a, b, c) \
  __builtin_amdgcn_wmma_f32_16x16x32_bf16(false, (a), false, (b), (short)0, (c), false, false)

// 16-byte async copy global -> LDS (gfx1250 GLOBAL_LOAD_ASYNC_TO_LDS_B128).
// lds = wave-relative LDS byte address (flat addr truncated to 32 bits),
// g   = 64-bit global address.  Tracked by ASYNCcnt.
__device__ __forceinline__ void async_b128(unsigned lds, const bf16* g) {
  asm volatile("global_load_async_to_lds_b128 %0, %1, off"
               :: "v"(lds), "v"(g)
               : "memory");
}
__device__ __forceinline__ unsigned lds_addr(const void* p) {
  return (unsigned)(unsigned long long)p;   // aperture: LDS address is the low 32 bits
}

// Load a 16x32 bf16 fragment following the CDNA5 16-bit A/B layout:
// lane&15 selects the row (M for A-frags, N for B-frags when the B-source is
// stored transposed, i.e. W is [N,K] row-major); half-wave selects K 0-7/16-23
// vs 8-15/24-31.  `p` points at the (row0, k0) tile origin, `ld` is the row
// stride in elements.  Works for both LDS and global pointers.
__device__ __forceinline__ v16bf load_frag(const bf16* p, int ld) {
  const int lane = threadIdx.x & 31;
  const bf16* q = p + (lane & 15) * ld + ((lane >> 4) << 3);
  v8bf lo = *(const v8bf*)(q);
  v8bf hi = *(const v8bf*)(q + 16);
  return __builtin_shufflevector(lo, hi, 0,1,2,3,4,5,6,7,8,9,10,11,12,13,14,15);
}

__device__ __forceinline__ float wred_add(float v) {
#pragma unroll
  for (int off = 16; off; off >>= 1) v += __shfl_xor(v, off, 32);
  return v;
}

// ---------------------------------------------------------------------------
// Generic WMMA GEMM:  C[M,N] = A[M,K] @ W[N,K]^T + bias
// A, W bf16 row-major; 128x64 block tile, 4 waves, each wave 32x64 (8 frags).
// Double-buffered async LDS staging: 6 async b128 per thread per K-step.
// OUT_BF16: store bf16 into Cb; ACC_ADD: Cf += result (residual fuse).
// ---------------------------------------------------------------------------
template <int OUT_BF16, int ACC_ADD>
__global__ __launch_bounds__(128) void gemm_bf16(
    const bf16* __restrict__ A, const bf16* __restrict__ W,
    const float* __restrict__ bias, float* __restrict__ Cf,
    bf16* __restrict__ Cb, int M, int N, int K)
{
  __shared__ alignas(16) bf16 As[2][128][40];
  __shared__ alignas(16) bf16 Ws[2][64][40];
  const int t = threadIdx.x, wave = t >> 5, lane = t & 31;
  const int m0 = blockIdx.y * 128, n0 = blockIdx.x * 64;
  v8f acc[2][4] = {};

  // Per-thread staging targets (constant across K-steps except buffer parity).
  auto issue = [&](int k0, int pbuf) {
    const bf16* asrc = A + (long)(m0 + t) * K + k0;       // row t, 32 elems
    const unsigned al = lds_addr(&As[pbuf][t][0]);
    async_b128(al,      asrc);
    async_b128(al + 16, asrc + 8);
    async_b128(al + 32, asrc + 16);
    async_b128(al + 48, asrc + 24);
    const bf16* wsrc = W + (long)(n0 + (t >> 1)) * K + k0 + (t & 1) * 16;
    const unsigned wl = lds_addr(&Ws[pbuf][t >> 1][(t & 1) * 16]);
    async_b128(wl,      wsrc);
    async_b128(wl + 16, wsrc + 8);
  };

  issue(0, 0);
  int pb = 0;
  for (int k0 = 0; k0 < K; k0 += 32) {
    if (k0 + 32 < K) {
      issue(k0 + 32, pb ^ 1);                 // fill other buffer
      asm volatile("s_wait_asynccnt 6" ::: "memory");   // cur batch landed
    } else {
      asm volatile("s_wait_asynccnt 0" ::: "memory");
    }
    __syncthreads();                          // all waves' copies visible

    v16bf bfr[4];
#pragma unroll
    for (int ni = 0; ni < 4; ++ni) bfr[ni] = load_frag(&Ws[pb][ni * 16][0], 40);
#pragma unroll
    for (int mi = 0; mi < 2; ++mi) {
      v16bf afr = load_frag(&As[pb][wave * 32 + mi * 16][0], 40);
#pragma unroll
      for (int ni = 0; ni < 4; ++ni) acc[mi][ni] = WMMA(afr, bfr[ni], acc[mi][ni]);
    }
    __syncthreads();                          // done reading before overwrite
    pb ^= 1;
  }

  // Epilogue.  D layout: lane&15 = N column, (lane>>4)*8 = M row offset.
  const int n = lane & 15, mo = (lane >> 4) << 3;
#pragma unroll
  for (int mi = 0; mi < 2; ++mi)
#pragma unroll
    for (int ni = 0; ni < 4; ++ni) {
      const int gn = n0 + ni * 16 + n;
      const float bv = bias ? bias[gn] : 0.0f;
#pragma unroll
      for (int r = 0; r < 8; ++r) {
        const long gm = m0 + wave * 32 + mi * 16 + mo + r;
        const float val = acc[mi][ni][r] + bv;
        if (ACC_ADD)        Cf[gm * N + gn] += val;
        else if (OUT_BF16)  Cb[gm * N + gn] = f2b(val);
        else                Cf[gm * N + gn] = val;
      }
    }
}

// ---------------------------------------------------------------------------
// Attention scores + rel-pos gathers + masked softmax -> bf16 probs.
// Block = (16 q-rows, head h, batch b); 8 waves each own 64 key columns.
// qk: [S*B, 2H] bf16 (row = s*B+b; q at col h*64, k at col 768+h*64)
// qp/kp: [S*B, NH, 64] f32 rel-pos dot tables; probs: [B,NH,S,S] bf16.
// ---------------------------------------------------------------------------
__global__ __launch_bounds__(256) void attn_scores_softmax(
    const bf16* __restrict__ qk, const float* __restrict__ qp,
    const float* __restrict__ kp, const int* __restrict__ pidx,
    const unsigned char* __restrict__ mask, bf16* __restrict__ probs)
{
  __shared__ float sc[16][S_];
  const int b = blockIdx.z, h = blockIdx.y, q0 = blockIdx.x * 16;
  const int wave = threadIdx.x >> 5, lane = threadIdx.x & 31;
  const int ldqk = B_ * 2 * H_;  // row step between consecutive seq positions

  v8f acc[4] = {};
  const bf16* qbase = qk + ((long)q0 * B_ + b) * (2 * H_) + h * HD_;
#pragma unroll
  for (int k0 = 0; k0 < HD_; k0 += 32) {
    v16bf a = load_frag(qbase + k0, ldqk);
#pragma unroll
    for (int ni = 0; ni < 4; ++ni) {
      const int col0 = wave * 64 + ni * 16;
      const bf16* kb = qk + ((long)col0 * B_ + b) * (2 * H_) + H_ + h * HD_ + k0;
      v16bf bb = load_frag(kb, ldqk);
      acc[ni] = WMMA(a, bb, acc[ni]);
    }
  }

  const int n = lane & 15, mo = (lane >> 4) << 3;
#pragma unroll
  for (int ni = 0; ni < 4; ++ni) {
    const int kg = wave * 64 + ni * 16 + n;
    const bool msk = mask[b * S_ + kg] != 0;
#pragma unroll
    for (int r = 0; r < 8; ++r) {
      const int ql = mo + r, qg = q0 + ql;
      const int bucket = pidx[qg * S_ + kg];
      float sv = acc[ni][r]
               + qp[(((long)qg * B_ + b) * NH_ + h) * 64 + bucket]
               + kp[(((long)kg * B_ + b) * NH_ + h) * 64 + bucket];
      sv *= SCALE_;
      sc[ql][kg] = msk ? -3.0e38f : sv;
    }
  }
  __syncthreads();

  // Row softmax: 16 threads per row, interleaved columns.
  const int row = threadIdx.x >> 4, j = threadIdx.x & 15;
  float mx = -3.0e38f;
  for (int i = j; i < S_; i += 16) mx = fmaxf(mx, sc[row][i]);
#pragma unroll
  for (int off = 8; off; off >>= 1) mx = fmaxf(mx, __shfl_xor(mx, off, 16));
  float sum = 0.f;
  for (int i = j; i < S_; i += 16) sum += __expf(sc[row][i] - mx);
#pragma unroll
  for (int off = 8; off; off >>= 1) sum += __shfl_xor(sum, off, 16);
  const float inv = 1.0f / sum;
  bf16* out = probs + (((long)(b * NH_ + h)) * S_ + q0 + row) * S_;
  for (int i = j; i < S_; i += 16) out[i] = f2b(__expf(sc[row][i] - mx) * inv);
}

// ---------------------------------------------------------------------------
// ctx = probs @ V.  Block = (64 q-rows, h, b); 4 waves, wave = 16 q-rows.
// V tile (32 k x 64 d) is transposed through LDS so B-frags read K-contiguous.
// ---------------------------------------------------------------------------
__global__ __launch_bounds__(128) void attn_ctx(
    const bf16* __restrict__ probs, const bf16* __restrict__ v,
    bf16* __restrict__ ctx)
{
  __shared__ alignas(16) bf16 VsT[HD_][40];   // [d][kk_local]
  const int b = blockIdx.z, h = blockIdx.y, q0 = blockIdx.x * 64;
  const int t = threadIdx.x, wave = t >> 5, lane = t & 31;
  v8f acc[4] = {};
  const bf16* arow = probs + (((long)(b * NH_ + h)) * S_ + q0 + wave * 16) * S_;

  for (int kk0 = 0; kk0 < S_; kk0 += 32) {
    const int kkl = t >> 2, db = (t & 3) << 4;
    const bf16* src = v + ((long)(kk0 + kkl) * B_ + b) * H_ + h * HD_ + db;
    v8bf c0 = *(const v8bf*)(src);
    v8bf c1 = *(const v8bf*)(src + 8);
    __syncthreads();
#pragma unroll
    for (int i = 0; i < 8; ++i) {
      VsT[db + i][kkl]     = c0[i];
      VsT[db + 8 + i][kkl] = c1[i];
    }
    __syncthreads();
    v16bf a = load_frag(arow + kk0, S_);
#pragma unroll
    for (int ni = 0; ni < 4; ++ni) {
      v16bf bb = load_frag(&VsT[ni * 16][0], 40);
      acc[ni] = WMMA(a, bb, acc[ni]);
    }
  }

  const int n = lane & 15, mo = (lane >> 4) << 3;
#pragma unroll
  for (int ni = 0; ni < 4; ++ni)
#pragma unroll
    for (int r = 0; r < 8; ++r) {
      const int qg = q0 + wave * 16 + mo + r;
      ctx[((long)qg * B_ + b) * H_ + h * HD_ + ni * 16 + n] = f2b(acc[ni][r]);
    }
}

// ---------------------------------------------------------------------------
// Elementwise / LayerNorm kernels (one wave per row, wave32 shuffles).
// ---------------------------------------------------------------------------
__global__ __launch_bounds__(256) void ln_rows(const float* __restrict__ in,
                                               bf16* __restrict__ out, int cols)
{
  const int wave = threadIdx.x >> 5, lane = threadIdx.x & 31;
  const long row = (long)blockIdx.x * 8 + wave;
  const float* p = in + row * cols;
  float s = 0.f, s2 = 0.f;
  for (int c = lane; c < cols; c += 32) { float v = p[c]; s += v; s2 += v * v; }
  s = wred_add(s); s2 = wred_add(s2);
  const float mu = s / cols;
  const float rs = rsqrtf(fmaxf(s2 / cols - mu * mu, 0.f) + 1e-7f);
  bf16* o = out + row * cols;
  for (int c = lane; c < cols; c += 32) o[c] = f2b((p[c] - mu) * rs);
}

__global__ __launch_bounds__(256) void embed_ln(const int* __restrict__ ids,
    const float* __restrict__ emb, float* __restrict__ x)
{
  const int wave = threadIdx.x >> 5, lane = threadIdx.x & 31;
  const int row = blockIdx.x * 8 + wave;          // row = s*B + b
  const int s = row >> 4, b = row & 15;
  const float* p = emb + (long)ids[b * S_ + s] * H_;
  float sm = 0.f, s2 = 0.f;
  for (int c = lane; c < H_; c += 32) { float v = p[c]; sm += v; s2 += v * v; }
  sm = wred_add(sm); s2 = wred_add(s2);
  const float mu = sm / H_;
  const float rs = rsqrtf(fmaxf(s2 / H_ - mu * mu, 0.f) + 1e-7f);
  float* o = x + (long)row * H_;
  for (int c = lane; c < H_; c += 32) o[c] = (p[c] - mu) * rs;
}

__global__ __launch_bounds__(256) void rel_ln(const float* __restrict__ emb,
    const float* __restrict__ g, const float* __restrict__ bta,
    float* __restrict__ out)
{
  const int wave = threadIdx.x >> 5, lane = threadIdx.x & 31;
  const int j = blockIdx.x * 8 + wave;
  if (j >= 63) return;
  const float* p = emb + (long)j * H_;
  float sm = 0.f, s2 = 0.f;
  for (int c = lane; c < H_; c += 32) { float v = p[c]; sm += v; s2 += v * v; }
  sm = wred_add(sm); s2 = wred_add(s2);
  const float mu = sm / H_;
  const float rs = rsqrtf(fmaxf(s2 / H_ - mu * mu, 0.f) + 1e-7f);
  float* o = out + (long)j * H_;
  for (int c = lane; c < H_; c += 32) o[c] = (p[c] - mu) * rs * g[c] + bta[c];
}

__global__ __launch_bounds__(256) void ln_aff_res(const float* __restrict__ tin,
    const float* __restrict__ g, const float* __restrict__ bta,
    float* __restrict__ x)
{
  const int wave = threadIdx.x >> 5, lane = threadIdx.x & 31;
  const long row = (long)blockIdx.x * 8 + wave;
  const float* p = tin + row * H_;
  float sm = 0.f, s2 = 0.f;
  for (int c = lane; c < H_; c += 32) { float v = p[c]; sm += v; s2 += v * v; }
  sm = wred_add(sm); s2 = wred_add(s2);
  const float mu = sm / H_;
  const float rs = rsqrtf(fmaxf(s2 / H_ - mu * mu, 0.f) + 1e-7f);
  float* o = x + row * H_;
  for (int c = lane; c < H_; c += 32) o[c] += (p[c] - mu) * rs * g[c] + bta[c];
}

// posp[j, 0:1536] = LN'd rel row j @ qkw_l^T + qkb_l   (tiny: 63x1536x768)
__global__ __launch_bounds__(256) void pos_proj(const float* __restrict__ rel,
    const float* __restrict__ w, const float* __restrict__ bias,
    float* __restrict__ out)
{
  const int id = blockIdx.x * 256 + threadIdx.x;
  if (id >= 63 * 1536) return;
  const int j = id / 1536, nn = id - j * 1536;
  const float* r = rel + (long)j * H_;
  const float* wr = w + (long)nn * H_;
  float acc = bias[nn];
  for (int c = 0; c < H_; ++c) acc += r[c] * wr[c];
  out[(long)j * 1536 + nn] = acc;
}

// qp[row,h,j] = q_row_h . kposTab[j,h];  kp[row,h,j] = k_row_h . qposTab[j,h]
__global__ __launch_bounds__(256) void qp_kp(const bf16* __restrict__ qk,
    const float* __restrict__ posp, float* __restrict__ qp,
    float* __restrict__ kp)
{
  const long id = (long)blockIdx.x * 256 + threadIdx.x;  // (row*12+h)*64 + j
  const int j = (int)(id & 63);
  if (j >= 63) return;
  const long rh = id >> 6;
  const int h = (int)(rh % NH_);
  const long row = rh / NH_;
  const bf16* qrow = qk + row * (2 * H_) + h * HD_;
  const bf16* krow = qrow + H_;
  const float* kpos = posp + (long)j * (2 * H_) + H_ + h * HD_;
  const float* qpos = posp + (long)j * (2 * H_) + h * HD_;
  float aq = 0.f, ak = 0.f;
#pragma unroll 8
  for (int d = 0; d < HD_; ++d) {
    aq += b2f(qrow[d]) * kpos[d];
    ak += b2f(krow[d]) * qpos[d];
  }
  qp[id] = aq;
  kp[id] = ak;
}

__global__ __launch_bounds__(256) void geglu(const float* __restrict__ y,
                                             float* __restrict__ z)
{
  const long id = (long)blockIdx.x * 256 + threadIdx.x;
  if (id >= (long)S_ * B_ * II_) return;
  const long row = id / II_;
  const int i = (int)(id - row * II_);
  const float a = y[row * (2 * II_) + i];
  const float g = y[row * (2 * II_) + II_ + i];
  const float t = tanhf(0.7978845608028654f * (g + 0.044715f * g * g * g));
  z[id] = a * (0.5f * g * (1.0f + t));
}

__global__ __launch_bounds__(256) void cvt_bf16(const float* __restrict__ in,
                                                bf16* __restrict__ out, long n)
{
  const long id = (long)blockIdx.x * 256 + threadIdx.x;
  if (id < n) out[id] = f2b(in[id]);
}

// ---------------------------------------------------------------------------
extern "C" void kernel_launch(void* const* d_in, const int* in_sizes, int n_in,
                              void* d_out, int out_size, void* d_ws,
                              size_t ws_size, hipStream_t stream)
{
  (void)in_sizes; (void)n_in; (void)out_size; (void)ws_size;
  const int*   ids  = (const int*)d_in[0];
  const unsigned char* mask = (const unsigned char*)d_in[1];
  const int*   pidx = (const int*)d_in[2];
  const float* word = (const float*)d_in[3];
  const float* remb = (const float*)d_in[4];
  const float* rg   = (const float*)d_in[5];
  const float* rb   = (const float*)d_in[6];
  const float* qkw  = (const float*)d_in[7];
  const float* qkb  = (const float*)d_in[8];
  const float* vw   = (const float*)d_in[9];
  const float* vb   = (const float*)d_in[10];
  const float* ow   = (const float*)d_in[11];
  const float* ob   = (const float*)d_in[12];
  const float* pg   = (const float*)d_in[13];
  const float* pb   = (const float*)d_in[14];
  const float* w1   = (const float*)d_in[15];
  const float* w2   = (const float*)d_in[16];

  char* wp = (char*)d_ws;
  auto alloc = [&](size_t bytes) -> void* {
    void* p = (void*)wp;
    wp += (bytes + 255) & ~(size_t)255;
    return p;
  };

  const long MR = (long)S_ * B_;  // 8192 token rows
  bf16*  wqk   = (bf16*)alloc((long)L_ * 1536 * 768 * 2);
  bf16*  wv    = (bf16*)alloc((long)L_ * 768 * 768 * 2);
  bf16*  wo    = (bf16*)alloc((long)L_ * 768 * 768 * 2);
  bf16*  ww1   = (bf16*)alloc((long)L_ * 4096 * 768 * 2);
  bf16*  ww2   = (bf16*)alloc((long)L_ * 768 * 2048 * 2);
  float* x     = (float*)alloc(MR * H_ * 4);
  bf16*  hb    = (bf16*)alloc(MR * H_ * 2);
  bf16*  qkbuf = (bf16*)alloc(MR * 1536 * 2);
  bf16*  vbuf  = (bf16*)alloc(MR * H_ * 2);
  float* relLN = (float*)alloc(64L * H_ * 4);
  float* posp  = (float*)alloc(64L * 1536 * 4);
  float* qpT   = (float*)alloc(MR * NH_ * 64 * 4);
  float* kpT   = (float*)alloc(MR * NH_ * 64 * 4);
  bf16*  probs = (bf16*)alloc((long)B_ * NH_ * S_ * S_ * 2);
  bf16*  ctxb  = (bf16*)alloc(MR * H_ * 2);
  float* tmp   = (float*)alloc(MR * H_ * 4);
  float* ybuf  = (float*)alloc(MR * I2_ * 4);
  float* zbuf  = (float*)alloc(MR * II_ * 4);
  bf16*  zb    = (bf16*)alloc(MR * II_ * 2);

  auto cvt = [&](const float* in, bf16* out, long n) {
    cvt_bf16<<<(int)((n + 255) / 256), 256, 0, stream>>>(in, out, n);
  };
  cvt(qkw, wqk, (long)L_ * 1536 * 768);
  cvt(vw,  wv,  (long)L_ * 768 * 768);
  cvt(ow,  wo,  (long)L_ * 768 * 768);
  cvt(w1,  ww1, (long)L_ * 4096 * 768);
  cvt(w2,  ww2, (long)L_ * 768 * 2048);

  embed_ln<<<1024, 256, 0, stream>>>(ids, word, x);
  rel_ln<<<8, 256, 0, stream>>>(remb, rg, rb, relLN);

  for (int l = 0; l < L_; ++l) {
    const bf16* Wqk = wqk + (long)l * 1536 * 768;
    const bf16* Wv  = wv  + (long)l * 768 * 768;
    const bf16* Wo  = wo  + (long)l * 768 * 768;
    const bf16* W1l = ww1 + (long)l * 4096 * 768;
    const bf16* W2l = ww2 + (long)l * 768 * 2048;

    ln_rows<<<1024, 256, 0, stream>>>(x, hb, H_);
    gemm_bf16<1, 0><<<dim3(24, 64), 128, 0, stream>>>(
        hb, Wqk, qkb + (long)l * 1536, nullptr, qkbuf, 8192, 1536, 768);
    gemm_bf16<1, 0><<<dim3(12, 64), 128, 0, stream>>>(
        hb, Wv, vb + (long)l * 768, nullptr, vbuf, 8192, 768, 768);

    pos_proj<<<(63 * 1536 + 255) / 256, 256, 0, stream>>>(
        relLN, qkw + (long)l * 1536 * 768, qkb + (long)l * 1536, posp);
    qp_kp<<<(int)(MR * NH_ * 64 / 256), 256, 0, stream>>>(qkbuf, posp, qpT, kpT);

    attn_scores_softmax<<<dim3(32, NH_, B_), 256, 0, stream>>>(
        qkbuf, qpT, kpT, pidx, mask, probs);
    attn_ctx<<<dim3(8, NH_, B_), 128, 0, stream>>>(probs, vbuf, ctxb);

    gemm_bf16<0, 0><<<dim3(12, 64), 128, 0, stream>>>(
        ctxb, Wo, ob + (long)l * 768, tmp, nullptr, 8192, 768, 768);
    ln_aff_res<<<1024, 256, 0, stream>>>(tmp, pg + (long)l * 768,
                                         pb + (long)l * 768, x);

    ln_rows<<<1024, 256, 0, stream>>>(x, hb, H_);
    gemm_bf16<0, 0><<<dim3(64, 64), 128, 0, stream>>>(
        hb, W1l, nullptr, ybuf, nullptr, 8192, 4096, 768);
    geglu<<<(int)((MR * II_ + 255) / 256), 256, 0, stream>>>(ybuf, zbuf);
    ln_rows<<<1024, 256, 0, stream>>>(zbuf, zb, II_);
    gemm_bf16<0, 1><<<dim3(12, 64), 128, 0, stream>>>(
        zb, W2l, nullptr, x, nullptr, 8192, 768, 2048);
  }

  hipMemcpyAsync(d_out, x, MR * H_ * 4, hipMemcpyDeviceToDevice, stream);
}
